// SelfAttention_83872121356804
// MI455X (gfx1250) — compile-verified, matrix operand order checked by reading, not run on hardware
//
#include <hip/hip_runtime.h>
#include <hip/hip_bf16.h>

typedef __attribute__((ext_vector_type(16))) _Float16 v16h;
typedef __attribute__((ext_vector_type(8)))  _Float16 v8h;
typedef __attribute__((ext_vector_type(8)))  float    v8f;

#define B_  4
#define N_  4096
#define C_  512
#define D_  64
#define G_  4            // split-K groups over the key dimension

static __device__ __forceinline__ v8f wmma_f16(v16h a, v16h b, v8f c) {
  // (neg_a, A, neg_b, B, c_mod, C, reuse_a, reuse_b)
  return __builtin_amdgcn_wmma_f32_16x16x32_f16(false, a, false, b, (short)0, c, false, false);
}

// Load one 16-element half-fragment: two contiguous 8-half (16B) runs at p and p+16.
static __device__ __forceinline__ v16h load_frag_h(const _Float16* p) {
  v16h f;
#pragma unroll
  for (int i = 0; i < 8; ++i) { f[i] = p[i]; f[8 + i] = p[16 + i]; }
  return f;
}

// ---------------------------------------------------------------------------
// Prep 1: x (fp32) -> xh (f16), 8 elements/thread, b128 in / b128 out.
// ---------------------------------------------------------------------------
__global__ __launch_bounds__(256) void cvt_x_kernel(const float* __restrict__ x,
                                                    _Float16* __restrict__ xh) {
  size_t i = ((size_t)blockIdx.x * 256 + threadIdx.x) * 8;
  const float4* p = (const float4*)(x + i);
  float4 a = p[0], b = p[1];
  v8h h;
  h[0] = (_Float16)a.x; h[1] = (_Float16)a.y; h[2] = (_Float16)a.z; h[3] = (_Float16)a.w;
  h[4] = (_Float16)b.x; h[5] = (_Float16)b.y; h[6] = (_Float16)b.z; h[7] = (_Float16)b.w;
  *(v8h*)(xh + i) = h;
}

// ---------------------------------------------------------------------------
// Prep 2: pack Wq/Wk/Wv (C x D fp32, row-major) -> WT[which][d][c] f16.
// ---------------------------------------------------------------------------
__global__ __launch_bounds__(256) void pack_w_kernel(const float* __restrict__ Wq,
                                                     const float* __restrict__ Wk,
                                                     const float* __restrict__ Wv,
                                                     _Float16* __restrict__ WT) {
  int t = blockIdx.x * 256 + threadIdx.x;          // 0 .. 3*D_*C_-1
  int w = t / (D_ * C_);
  int r = t - w * (D_ * C_);
  int d = r / C_;
  int c = r - d * C_;
  const float* W = (w == 0) ? Wq : (w == 1) ? Wk : Wv;
  WT[t] = (_Float16)W[c * D_ + d];                 // coalesced store
}

// ---------------------------------------------------------------------------
// Kernel 1: fused QKV projection. grid = (B*N/16, 3), block = 128.
// ---------------------------------------------------------------------------
template <bool XH>
__global__ __launch_bounds__(128) void qkv_proj_kernel(
    const float* __restrict__ x, const _Float16* __restrict__ xh,
    const _Float16* __restrict__ WT,
    const float* __restrict__ bq, const float* __restrict__ bk,
    const float* __restrict__ bv,
    _Float16* __restrict__ qws, _Float16* __restrict__ kws,
    _Float16* __restrict__ vTws)
{
  const int wave  = threadIdx.x >> 5;
  const int lane  = threadIdx.x & 31;
  const int rt    = blockIdx.x;      // 16-row tile over B*N rows
  const int which = blockIdx.y;      // 0 = Q, 1 = K, 2 = V
  const int d0 = wave * 16;
  const int m  = lane & 15;
  const int hi = lane >> 4;
  const int kb = hi * 8;

  const float* bias = (which == 0) ? bq : (which == 1) ? bk : bv;
  const _Float16* wrow = WT + ((size_t)which * D_ + d0 + m) * C_;

  const int row = rt * 16 + m;
  v8f c0 = {0.f, 0.f, 0.f, 0.f, 0.f, 0.f, 0.f, 0.f};
  v8f c1 = c0;

#pragma unroll
  for (int kt4 = 0; kt4 < C_ / 32; kt4 += 4) {
    v16h af[4], bf[4];
#pragma unroll
    for (int i = 0; i < 4; ++i) {
      const int kt = kt4 + i;
      if (XH) {
        af[i] = load_frag_h(xh + (size_t)row * C_ + kt * 32 + kb);
      } else {
        const float* pa = x + (size_t)row * C_ + kt * 32 + kb;
#pragma unroll
        for (int e = 0; e < 8; ++e) {
          af[i][e] = (_Float16)pa[e]; af[i][8 + e] = (_Float16)pa[16 + e];
        }
      }
      bf[i] = load_frag_h(wrow + kt * 32 + kb);   // B[k][n] = WT[which][d0+n][k]
    }
    c0 = wmma_f16(af[0], bf[0], c0);
    c1 = wmma_f16(af[1], bf[1], c1);
    c0 = wmma_f16(af[2], bf[2], c0);
    c1 = wmma_f16(af[3], bf[3], c1);
  }
#pragma unroll
  for (int r = 0; r < 8; ++r) c0[r] += c1[r];

  const float bias_n = bias[d0 + m];
  if (which < 2) {
    const float scl = (which == 0) ? 0.125f : 1.0f;  // fold dk^-0.5 into Q
    _Float16* dst = (which == 0) ? qws : kws;
#pragma unroll
    for (int r = 0; r < 8; ++r) {
      int ro = rt * 16 + r + hi * 8;                 // C-layout: VGPR r -> rows r, r+8
      dst[(size_t)ro * D_ + d0 + m] = (_Float16)((c0[r] + bias_n) * scl);
    }
  } else {
    // V transposed: vT[b][d][n] so PV B-fragments are contiguous along keys
#pragma unroll
    for (int r = 0; r < 8; ++r) {
      int ro   = rt * 16 + r + hi * 8;
      int bidx = ro >> 12;                            // / N_
      int nloc = ro & (N_ - 1);
      vTws[((size_t)bidx * D_ + d0 + m) * N_ + nloc] = (_Float16)(c0[r] + bias_n);
    }
  }
}

// ---------------------------------------------------------------------------
// Kernel 2: flash attention, software-pipelined K (ping-pong regs).
// SPLIT=true: grid (B*N/64, G_), each wave covers N_/G_ keys, writes partials.
// SPLIT=false: grid (B*N/64), full key range, writes final output.
// ---------------------------------------------------------------------------
template <bool SPLIT>
__global__ __launch_bounds__(128) void flash_attn_kernel(
    const _Float16* __restrict__ qws, const _Float16* __restrict__ kws,
    const _Float16* __restrict__ vTws, float* __restrict__ out,
    float* __restrict__ pacc, float* __restrict__ pm, float* __restrict__ pl)
{
  const int LP = 72;                                  // padded row stride (halves)
  __shared__ __align__(16) _Float16 plds[4][16 * LP]; // per-wave P staging tile

  const int wave = threadIdx.x >> 5;
  const int lane = threadIdx.x & 31;
  const int qt = blockIdx.x * 4 + wave;   // global 16-query tile (0..1023)
  const int g  = SPLIT ? blockIdx.y : 0;  // key group
  const int b  = qt >> 8;                 // 256 q-tiles per batch
  const int q0 = qt * 16;                 // global row over B*N
  const int m  = lane & 15;
  const int hi = lane >> 4;
  const int kb = hi * 8;

  const int jBeg = SPLIT ? g * (N_ / G_) : 0;
  const int jEnd = SPLIT ? jBeg + (N_ / G_) : N_;

  _Float16* P = plds[wave];
  const _Float16* vrow = vTws + (size_t)b * D_ * N_;
  const _Float16* krow = kws + (size_t)b * N_ * D_;

  // Q fragments: two 16x32 A-tiles covering head dims 0..31 and 32..63
  v16h qa0 = load_frag_h(qws + (size_t)(q0 + m) * D_ + 0  + kb);
  v16h qa1 = load_frag_h(qws + (size_t)(q0 + m) * D_ + 32 + kb);

  v8f acc[4];
#pragma unroll
  for (int t = 0; t < 4; ++t)
#pragma unroll
    for (int r = 0; r < 8; ++r) acc[t][r] = 0.f;

  float rm[8], rl[8];
#pragma unroll
  for (int r = 0; r < 8; ++r) { rm[r] = -1e30f; rl[r] = 0.f; }

  // Load 8 K fragments (64 keys) for key offset jj into kf[].
  auto load_k = [&](v16h* kf, int jj) {
    const _Float16* kp = krow + (size_t)jj * D_;
#pragma unroll
    for (int t = 0; t < 4; ++t) {
      kf[2 * t]     = load_frag_h(kp + (t * 16 + m) * D_ + kb);       // dims 0..31
      kf[2 * t + 1] = load_frag_h(kp + (t * 16 + m) * D_ + 32 + kb);  // dims 32..63
    }
  };

  // One 64-key flash step: consume kfc (keys jc), refill kfn (keys jn).
  auto step = [&](v16h* kfc, int jc, v16h* kfn, int jn) {
    // --- S = Q * K^T for 4 sub-tiles of 16 keys (scale pre-folded into Q) ---
    v8f s[4];
#pragma unroll
    for (int t = 0; t < 4; ++t) {
      v8f z = {0.f, 0.f, 0.f, 0.f, 0.f, 0.f, 0.f, 0.f};
      s[t] = wmma_f16(qa0, kfc[2 * t], z);
      s[t] = wmma_f16(qa1, kfc[2 * t + 1], s[t]);
    }

    // --- Issue V-fragment loads; latency hides under softmax VALU work ---
    v16h vf[8];
    const _Float16* vbase = vrow + jc + kb;
#pragma unroll
    for (int nt = 0; nt < 4; ++nt) {
      const _Float16* vp = vbase + (size_t)(nt * 16 + m) * N_;
      vf[2 * nt]     = load_frag_h(vp);        // keys jc..jc+31
      vf[2 * nt + 1] = load_frag_h(vp + 32);   // keys jc+32..jc+63
    }

    // --- Pipeline: refill next K buffer, consumed a half-iteration later ---
    if (jn < jEnd) load_k(kfn, jn);

    // --- online softmax (rows r, r+8 live in VGPR r of halves 0/1) ---
    float alpha[8];
#pragma unroll
    for (int r = 0; r < 8; ++r) {
      float mx = fmaxf(fmaxf(s[0][r], s[1][r]), fmaxf(s[2][r], s[3][r]));
      mx = fmaxf(mx, __shfl_xor(mx, 1, 32));
      mx = fmaxf(mx, __shfl_xor(mx, 2, 32));
      mx = fmaxf(mx, __shfl_xor(mx, 4, 32));
      mx = fmaxf(mx, __shfl_xor(mx, 8, 32));
      float mnew = fmaxf(rm[r], mx);
      float a = __expf(rm[r] - mnew);
      rm[r] = mnew;
      alpha[r] = a;
      float p0 = __expf(s[0][r] - mnew);
      float p1 = __expf(s[1][r] - mnew);
      float p2 = __expf(s[2][r] - mnew);
      float p3 = __expf(s[3][r] - mnew);
      s[0][r] = p0; s[1][r] = p1; s[2][r] = p2; s[3][r] = p3;
      float sum = (p0 + p1) + (p2 + p3);
      sum += __shfl_xor(sum, 1, 32);
      sum += __shfl_xor(sum, 2, 32);
      sum += __shfl_xor(sum, 4, 32);
      sum += __shfl_xor(sum, 8, 32);
      rl[r] = rl[r] * a + sum;
    }
#pragma unroll
    for (int r = 0; r < 8; ++r) {
      acc[0][r] *= alpha[r]; acc[1][r] *= alpha[r];
      acc[2][r] *= alpha[r]; acc[3][r] *= alpha[r];
    }

    // --- P (C-layout f32) -> LDS f16 [row][k64], re-read as two A-fragments ---
#pragma unroll
    for (int r = 0; r < 8; ++r) {
      int prow = r + hi * 8;
#pragma unroll
      for (int t = 0; t < 4; ++t)
        P[prow * LP + t * 16 + m] = (_Float16)s[t][r];
    }
    asm volatile("s_wait_dscnt 0" ::: "memory");
    v16h pa0 = load_frag_h(P + m * LP + kb);        // keys jc..jc+31
    v16h pa1 = load_frag_h(P + m * LP + 32 + kb);   // keys jc+32..jc+63

    // --- O += P * V ---
#pragma unroll
    for (int nt = 0; nt < 4; ++nt) {
      acc[nt] = wmma_f16(pa0, vf[2 * nt], acc[nt]);
      acc[nt] = wmma_f16(pa1, vf[2 * nt + 1], acc[nt]);
    }
  };

  v16h kfA[8], kfB[8];
  load_k(kfA, jBeg);

  for (int j = jBeg; j < jEnd; j += 128) {
    // Prefetch the next 128-key K/V tiles into the WGP cache.
    if (j + 128 < jEnd) {
      const _Float16* pk = krow + (size_t)(j + 128) * D_;
      __builtin_prefetch(pk + (size_t)lane * 128, 0, 3);
      __builtin_prefetch(pk + 4096 + (size_t)lane * 128, 0, 3);
      __builtin_prefetch(vrow + (size_t)(2 * lane) * N_ + j + 128, 0, 3);
      __builtin_prefetch(vrow + (size_t)(2 * lane + 1) * N_ + j + 128, 0, 3);
      __builtin_prefetch(vrow + (size_t)(2 * lane) * N_ + j + 192, 0, 3);
      __builtin_prefetch(vrow + (size_t)(2 * lane + 1) * N_ + j + 192, 0, 3);
    }
    step(kfA, j,       kfB, j + 64);
    step(kfB, j + 64,  kfA, j + 128);
  }

  if (SPLIT) {
    // Store unnormalized partial acc + per-row (m, l); combined later.
    const int BN = B_ * N_;
    float* pa = pacc + ((size_t)g * BN + q0) * D_;
#pragma unroll
    for (int r = 0; r < 8; ++r) {
      float* po = pa + (size_t)(r + hi * 8) * D_;
      po[0  + m] = acc[0][r];
      po[16 + m] = acc[1][r];
      po[32 + m] = acc[2][r];
      po[48 + m] = acc[3][r];
    }
#pragma unroll
    for (int r = 0; r < 8; ++r) {
      if (m == r) {                       // 2 lanes (hi=0/1) -> rows r, r+8
        pm[(size_t)g * BN + q0 + r + hi * 8] = rm[r];
        pl[(size_t)g * BN + q0 + r + hi * 8] = rl[r];
      }
    }
  } else {
    // Normalize and store (coalesced 16-float runs per row).
#pragma unroll
    for (int r = 0; r < 8; ++r) {
      float inv = 1.0f / rl[r];
      int ro = q0 + r + hi * 8;
      float* po = out + (size_t)ro * D_;
      po[0  + m] = acc[0][r] * inv;
      po[16 + m] = acc[1][r] * inv;
      po[32 + m] = acc[2][r] * inv;
      po[48 + m] = acc[3][r] * inv;
    }
  }
}

// ---------------------------------------------------------------------------
// Kernel 3: combine split-K partials. One thread per (row, d).
// ---------------------------------------------------------------------------
__global__ __launch_bounds__(256) void combine_kernel(
    const float* __restrict__ pacc, const float* __restrict__ pm,
    const float* __restrict__ pl, float* __restrict__ out)
{
  const int BN = B_ * N_;
  int t = blockIdx.x * 256 + threadIdx.x;   // 0 .. BN*D_-1
  int qrow = t >> 6;
  float M = -1e30f;
#pragma unroll
  for (int g = 0; g < G_; ++g) M = fmaxf(M, pm[(size_t)g * BN + qrow]);
  float L = 0.f, o = 0.f;
#pragma unroll
  for (int g = 0; g < G_; ++g) {
    float w = __expf(pm[(size_t)g * BN + qrow] - M);
    L += w * pl[(size_t)g * BN + qrow];
    o += w * pacc[(size_t)g * BN * D_ + t];
  }
  out[t] = o / L;
}

// ---------------------------------------------------------------------------
extern "C" void kernel_launch(void* const* d_in, const int* in_sizes, int n_in,
                              void* d_out, int out_size, void* d_ws, size_t ws_size,
                              hipStream_t stream) {
  (void)in_sizes; (void)n_in; (void)out_size;
  const float* x  = (const float*)d_in[0];
  const float* Wq = (const float*)d_in[1];
  const float* bq = (const float*)d_in[2];
  const float* Wk = (const float*)d_in[3];
  const float* bk = (const float*)d_in[4];
  const float* Wv = (const float*)d_in[5];
  const float* bv = (const float*)d_in[6];
  float* out = (float*)d_out;

  const size_t elems    = (size_t)B_ * N_ * D_;    // 1,048,576
  const size_t wt_elems = (size_t)3 * D_ * C_;     // 98,304
  const size_t x_elems  = (size_t)B_ * N_ * C_;    // 8,388,608

  // Workspace layout (halves):  qws | kws | vT | WT | xh || (floats) pacc | pm | pl
  _Float16* qws = (_Float16*)d_ws;
  _Float16* kws = qws + elems;
  _Float16* vT  = kws + elems;
  _Float16* WT  = vT + elems;
  _Float16* xh  = WT + wt_elems;

  const size_t half_bytes  = (3 * elems + wt_elems + x_elems) * sizeof(_Float16);
  float* pacc = (float*)((char*)d_ws + half_bytes);
  float* pm   = pacc + (size_t)G_ * B_ * N_ * D_;
  float* pl   = pm + (size_t)G_ * B_ * N_;
  const size_t split_bytes =
      half_bytes + ((size_t)G_ * B_ * N_ * (D_ + 2)) * sizeof(float);

  const bool use_xh    = ws_size >= half_bytes;
  const bool use_split = ws_size >= split_bytes;

  pack_w_kernel<<<(int)(wt_elems / 256), 256, 0, stream>>>(Wq, Wk, Wv, WT);
  if (use_xh) {
    cvt_x_kernel<<<(int)(x_elems / (256 * 8)), 256, 0, stream>>>(x, xh);
    qkv_proj_kernel<true><<<dim3((B_ * N_) / 16, 3), 128, 0, stream>>>(
        x, xh, WT, bq, bk, bv, qws, kws, vT);
  } else {
    qkv_proj_kernel<false><<<dim3((B_ * N_) / 16, 3), 128, 0, stream>>>(
        x, xh, WT, bq, bk, bv, qws, kws, vT);
  }

  if (use_split) {
    flash_attn_kernel<true><<<dim3((B_ * N_) / 64, G_), 128, 0, stream>>>(
        qws, kws, vT, out, pacc, pm, pl);
    combine_kernel<<<(B_ * N_ * D_) / 256, 256, 0, stream>>>(pacc, pm, pl, out);
  } else {
    flash_attn_kernel<false><<<(B_ * N_) / 64, 128, 0, stream>>>(
        qws, kws, vT, out, pacc, pm, pl);
  }
}
